// RWKV_Tmix_x070_42004780155476
// MI455X (gfx1250) — compile-verified, matrix operand order checked by reading, not run on hardware
//
#include <hip/hip_runtime.h>
#include <hip/hip_bf16.h>
#include <math.h>

#define Bsz 4
#define Tsz 2048
#define Csz 2048
#define Hh  32
#define HS  64

typedef __bf16 bf16_t;
typedef __attribute__((ext_vector_type(16))) __bf16 v16bf;
typedef __attribute__((ext_vector_type(8)))  __bf16 v8bf;
typedef __attribute__((ext_vector_type(8)))  float  v8f;
typedef __attribute__((ext_vector_type(4)))  float  f32x4;

__device__ __forceinline__ float sigmoidf_(float x) { return 1.f / (1.f + expf(-x)); }

// ---------- bulk f32 -> bf16 convert (8 elems/thread) ----------
__global__ __launch_bounds__(256) void cvt_f32_bf16(const float* __restrict__ in,
                                                    bf16_t* __restrict__ out) {
  size_t i = ((size_t)blockIdx.x * 256 + threadIdx.x) * 8;
  f32x4 a = *(const f32x4*)(in + i);
  f32x4 b = *(const f32x4*)(in + i + 4);
  v8bf o;
#pragma unroll
  for (int e = 0; e < 4; ++e) {
    o[e]     = (bf16_t)a[e];
    o[4 + e] = (bf16_t)b[e];
  }
  *(v8bf*)(out + i) = o;
}

// ---------- NT GEMM: C[m,n] = sum_k A[m,k]*B[n,k], pure bf16 WMMA loop ----------
// A: M x K bf16 row-major, Bm: N x K bf16 row-major (X @ W.T with row-major W).
// Block: 256 thr = 8 waves (2x4), block tile 128x256, wave tile 64x64 (4x4 frags).
__global__ __launch_bounds__(256, 1) void gemm_nt_wmma(
    const bf16_t* __restrict__ A, const bf16_t* __restrict__ Bm,
    float* __restrict__ Cm, int M, int N, int K) {
  const int lane = threadIdx.x & 31;
  const int wave = threadIdx.x >> 5;
  const int wm = wave >> 2;  // 0..1
  const int wn = wave & 3;   // 0..3
  const int m0 = blockIdx.x * 128 + wm * 64;
  const int n0 = blockIdx.y * 256 + wn * 64;
  const int row  = lane & 15;          // fragment row (M for A, N for B)
  const int koff = (lane >> 4) << 4;   // K half selected by lane[4]

  const bf16_t* Ab = A  + (size_t)(m0 + row) * K + koff;
  const bf16_t* Bb = Bm + (size_t)(n0 + row) * K + koff;

  v8f acc[4][4];
  {
    v8f z = {};
#pragma unroll
    for (int i = 0; i < 4; ++i)
#pragma unroll
      for (int j = 0; j < 4; ++j) acc[i][j] = z;
  }

  for (int k0 = 0; k0 < K; k0 += 32) {
    v16bf af[4], bfr[4];
#pragma unroll
    for (int i = 0; i < 4; ++i)
      af[i] = *(const v16bf*)(Ab + (size_t)(i * 16) * K + k0);
#pragma unroll
    for (int j = 0; j < 4; ++j)
      bfr[j] = *(const v16bf*)(Bb + (size_t)(j * 16) * K + k0);
    if (k0 + 256 < K) {  // gfx1250 global_prefetch_b8, 8 K-tiles ahead
      __builtin_prefetch(Ab + k0 + 256, 0, 3);
      __builtin_prefetch(Bb + k0 + 256, 0, 3);
    }
#pragma unroll
    for (int i = 0; i < 4; ++i)
#pragma unroll
      for (int j = 0; j < 4; ++j)
        acc[i][j] = __builtin_amdgcn_wmma_f32_16x16x32_bf16(
            false, af[i], false, bfr[j], (short)0, acc[i][j], false, false);
  }

  const int crow = (lane >> 4) << 3;
  const int ccol = lane & 15;
#pragma unroll
  for (int i = 0; i < 4; ++i)
#pragma unroll
    for (int j = 0; j < 4; ++j)
#pragma unroll
      for (int e = 0; e < 8; ++e)
        Cm[(size_t)(m0 + i * 16 + crow + e) * N + (n0 + j * 16 + ccol)] = acc[i][j][e];
}

// ---------- token shift + 6-way mix; GEMM operands emitted as bf16 ----------
__global__ __launch_bounds__(256) void mix_kernel(
    const float* __restrict__ x,
    const float* __restrict__ cr, const float* __restrict__ cw,
    const float* __restrict__ ck, const float* __restrict__ cv,
    const float* __restrict__ ca, const float* __restrict__ cg,
    bf16_t* __restrict__ xrb, float* __restrict__ xw, bf16_t* __restrict__ xkb,
    float* __restrict__ xv, bf16_t* __restrict__ xvb,
    float* __restrict__ xa, float* __restrict__ xg) {
  size_t idx = (size_t)blockIdx.x * 256 + threadIdx.x;
  int c = (int)(idx & (Csz - 1));
  size_t bt = idx >> 11;                // / Csz
  int t = (int)(bt & (Tsz - 1));
  float xc = x[idx];
  float xp = (t > 0) ? x[idx - Csz] : 0.f;
  float xx = xp - xc;
  float vr = xc + xx * cr[c];
  float vk = xc + xx * ck[c];
  float vv = xc + xx * cv[c];
  xrb[idx] = (bf16_t)vr;
  xkb[idx] = (bf16_t)vk;
  xvb[idx] = (bf16_t)vv;
  xv[idx]  = vv;
  xw[idx]  = xc + xx * cw[c];
  xa[idx]  = xc + xx * ca[c];
  xg[idx]  = xc + xx * cg[c];
}

// ---------- LoRA stage 1: out[m,d] = act( X[m,:] . W1[:,d] ), D<=128 ----------
__global__ void lora_stage1(const float* __restrict__ X, const float* __restrict__ W1,
                            float* __restrict__ out, int D, int act) {
  __shared__ float xrow[Csz];
  const int m = blockIdx.x;
  const int d = threadIdx.x;
  for (int kx = threadIdx.x; kx < Csz; kx += blockDim.x)
    xrow[kx] = X[(size_t)m * Csz + kx];
  __syncthreads();
  float s = 0.f;
  for (int kx = 0; kx < Csz; ++kx)
    s = fmaf(xrow[kx], W1[(size_t)kx * D + d], s);
  if (act == 1)      s = tanhf(s);
  else if (act == 2) s = sigmoidf_(s);
  out[(size_t)m * D + d] = s;
}

// ---------- LoRA stage 2: s = tmp[m,:] . W2[:,c]; mode-specific epilogue ----------
// mode 0: wdec = exp(-exp(-softplus(-(w0+s)) - 0.5))
// mode 1: a = sigmoid(a0+s)
// mode 2: v = v + (v_first - v) * sigmoid(v0+s)   (in-place on vin==out)
// mode 3: g = s
__global__ __launch_bounds__(256) void lora_stage2(
    const float* __restrict__ tmp, const float* __restrict__ W2,
    const float* __restrict__ bias, int D, int mode,
    const float* vin, const float* __restrict__ vfirst, float* out) {
  __shared__ float tl[128];
  const int m = blockIdx.x;
  const int c = blockIdx.y * 256 + threadIdx.x;
  if (threadIdx.x < D) tl[threadIdx.x] = tmp[(size_t)m * D + threadIdx.x];
  __syncthreads();
  float s = 0.f;
  for (int d = 0; d < D; ++d)
    s = fmaf(tl[d], W2[(size_t)d * Csz + c], s);
  size_t idx = (size_t)m * Csz + c;
  if (mode == 0) {
    float yv  = bias[c] + s;
    float nsp = fminf(yv, 0.f) - log1pf(expf(-fabsf(yv)));  // -softplus(-yv), stable
    out[idx] = expf(-expf(nsp - 0.5f));
  } else if (mode == 1) {
    out[idx] = sigmoidf_(bias[c] + s);
  } else if (mode == 2) {
    float sg = sigmoidf_(bias[c] + s);
    float vv = vin[idx];
    out[idx] = vv + (vfirst[idx] - vv) * sg;
  } else {
    out[idx] = s;
  }
}

// ---------- per-head k_k normalize + derived sequences (one wave per (b,t,h)) ----------
__global__ __launch_bounds__(256) void kknorm_kernel(
    const float* k, const float* __restrict__ a,
    const float* __restrict__ k_k, const float* __restrict__ k_a,
    float* __restrict__ kk, float* __restrict__ bb, float* knew) {
  int gid = blockIdx.x * 8 + (threadIdx.x >> 5);
  int lane = threadIdx.x & 31;
  int h = gid & (Hh - 1);
  size_t bt = (size_t)(gid >> 5);
  size_t base = bt * Csz + (size_t)h * HS;
  int c0 = h * HS + lane, c1 = c0 + 32;
  float k0v = k[base + lane], k1v = k[base + lane + 32];
  float a0v = a[base + lane], a1v = a[base + lane + 32];
  float q0 = k0v * k_k[c0], q1 = k1v * k_k[c1];
  float ss = q0 * q0 + q1 * q1;
#pragma unroll
  for (int mm = 1; mm < 32; mm <<= 1) ss += __shfl_xor(ss, mm, 32);
  float inv = 1.f / fmaxf(sqrtf(ss), 1e-12f);
  float kk0 = q0 * inv, kk1 = q1 * inv;
  kk[base + lane] = kk0;       kk[base + lane + 32] = kk1;
  bb[base + lane] = kk0 * a0v; bb[base + lane + 32] = kk1 * a1v;
  knew[base + lane]      = k0v * (1.f + (a0v - 1.f) * k_a[c0]);
  knew[base + lane + 32] = k1v * (1.f + (a1v - 1.f) * k_a[c1]);
}

// ---------- delta-rule recurrence: 1 block per (b,h), 4 lanes per state row ----------
__global__ __launch_bounds__(256) void rwkv_rec_kernel(
    const float* __restrict__ rp, const float* __restrict__ wp,
    const float* __restrict__ kp, const float* __restrict__ vp,
    const float* __restrict__ kkp, const float* __restrict__ bbp,
    float* __restrict__ y) {
  const int bh = blockIdx.x;
  const int b = bh / Hh, h = bh % Hh;
  const int tid = threadIdx.x;
  const int i = tid >> 2;       // state row (value index)
  const int p = tid & 3;        // column quarter
  const int j0 = p << 4;
  __shared__ float sh[6][HS];   // r, wdec, k, v, (-kk), bb
  float S[16];
#pragma unroll
  for (int q = 0; q < 16; ++q) S[q] = 0.f;
  const size_t base = ((size_t)b * Tsz) * Csz + (size_t)h * HS;

  for (int t = 0; t < Tsz; ++t) {
    size_t off = base + (size_t)t * Csz;
    {  // cooperative stage of 6x64 floats
      int u = tid, arr = u >> 6, e = u & 63;
      const float* s = (arr == 0) ? rp : (arr == 1) ? wp : (arr == 2) ? kp : vp;
      sh[arr][e] = s[off + e];
    }
    if (tid < 128) {
      int u = tid + 256, arr = u >> 6, e = u & 63;
      float val = (arr == 4) ? -kkp[off + e] : bbp[off + e];
      sh[arr][e] = val;
    }
    __syncthreads();

    float vi = sh[3][i];
    float sa = 0.f;
#pragma unroll
    for (int q = 0; q < 16; ++q) sa = fmaf(S[q], sh[4][j0 + q], sa);
    sa += __shfl_xor(sa, 1, 32);
    sa += __shfl_xor(sa, 2, 32);

    float op = 0.f;
#pragma unroll
    for (int q = 0; q < 16; ++q) {
      int j = j0 + q;
      float s2 = fmaf(S[q], sh[1][j], fmaf(sa, sh[5][j], vi * sh[2][j]));
      S[q] = s2;
      op = fmaf(s2, sh[0][j], op);
    }
    op += __shfl_xor(op, 1, 32);
    op += __shfl_xor(op, 2, 32);
    if (p == 0) y[off + i] = op;
    __syncthreads();
  }
}

// ---------- groupnorm + (r.k.r_k)*v residual + gate -> bf16 GEMM operand ----------
__global__ __launch_bounds__(256) void post_kernel(
    const float* __restrict__ y, const float* __restrict__ r,
    const float* __restrict__ k, const float* __restrict__ v,
    const float* __restrict__ g, const float* __restrict__ r_k,
    const float* __restrict__ ln_g, const float* __restrict__ ln_b,
    bf16_t* __restrict__ yg) {
  int gid = blockIdx.x * 8 + (threadIdx.x >> 5);
  int lane = threadIdx.x & 31;
  int h = gid & (Hh - 1);
  size_t bt = (size_t)(gid >> 5);
  size_t base = bt * Csz + (size_t)h * HS;
  int c0 = h * HS + lane, c1 = c0 + 32;
  float y0 = y[base + lane], y1 = y[base + lane + 32];
  float s1 = y0 + y1, s2 = y0 * y0 + y1 * y1;
  float rkp = r[base + lane]      * k[base + lane]      * r_k[c0]
            + r[base + lane + 32] * k[base + lane + 32] * r_k[c1];
#pragma unroll
  for (int mm = 1; mm < 32; mm <<= 1) {
    s1  += __shfl_xor(s1, mm, 32);
    s2  += __shfl_xor(s2, mm, 32);
    rkp += __shfl_xor(rkp, mm, 32);
  }
  float mu  = s1 * (1.f / 64.f);
  float var = s2 * (1.f / 64.f) - mu * mu;
  float inv = rsqrtf(var + 0.00064f);
  float n0 = (y0 - mu) * inv * ln_g[c0] + ln_b[c0] + rkp * v[base + lane];
  float n1 = (y1 - mu) * inv * ln_g[c1] + ln_b[c1] + rkp * v[base + lane + 32];
  yg[base + lane]      = (bf16_t)(n0 * g[base + lane]);
  yg[base + lane + 32] = (bf16_t)(n1 * g[base + lane + 32]);
}

// ---------- host-side orchestration ----------
extern "C" void kernel_launch(void* const* d_in, const int* in_sizes, int n_in,
                              void* d_out, int out_size, void* d_ws, size_t ws_size,
                              hipStream_t stream) {
  (void)in_sizes; (void)n_in; (void)out_size; (void)ws_size;
  const float* x      = (const float*)d_in[0];
  const float* vfirst = (const float*)d_in[1];
  const float* c_r = (const float*)d_in[2];
  const float* c_w = (const float*)d_in[3];
  const float* c_k = (const float*)d_in[4];
  const float* c_v = (const float*)d_in[5];
  const float* c_a = (const float*)d_in[6];
  const float* c_g = (const float*)d_in[7];
  const float* w0  = (const float*)d_in[8];
  const float* a0  = (const float*)d_in[9];
  const float* v0  = (const float*)d_in[10];
  const float* k_k = (const float*)d_in[11];
  const float* k_a = (const float*)d_in[12];
  const float* w1  = (const float*)d_in[13];
  const float* w2  = (const float*)d_in[14];
  const float* a1  = (const float*)d_in[15];
  const float* a2  = (const float*)d_in[16];
  const float* v1  = (const float*)d_in[17];
  const float* v2  = (const float*)d_in[18];
  const float* g1  = (const float*)d_in[19];
  const float* g2  = (const float*)d_in[20];
  const float* r_k = (const float*)d_in[21];
  const float* W_r = (const float*)d_in[22];
  const float* W_k = (const float*)d_in[23];
  const float* W_v = (const float*)d_in[24];
  const float* W_o = (const float*)d_in[25];
  const float* lng = (const float*)d_in[26];
  const float* lnb = (const float*)d_in[27];

  const size_t SL = (size_t)Bsz * Tsz * Csz;  // 16,777,216 elements per activation
  const size_t WN = (size_t)Csz * Csz;        // 4,194,304 weight elements
  const int M = Bsz * Tsz, N = Csz, K = Csz;
  float* ws = (float*)d_ws;
  // float slots (64 MB each)
  float* xw = ws + 0 * SL;  float* xa = ws + 1 * SL;  float* xv = ws + 2 * SL;
  float* xg = ws + 3 * SL;  float* rb = ws + 4 * SL;  float* wb = ws + 5 * SL;
  float* kb = ws + 6 * SL;  float* vb = ws + 7 * SL;  float* ab = ws + 8 * SL;
  float* gb = ws + 9 * SL;  float* tmp = ws + 10 * SL;
  float* tmp_w = tmp;
  float* tmp_a = tmp + (size_t)M * 64;
  float* tmp_v = tmp + (size_t)M * 128;
  float* tmp_g = tmp + (size_t)M * 160;
  // bf16 buffers: two activations per float slot, weight staging in slack
  bf16_t* xr_bf  = (bf16_t*)(ws + 11 * SL);          // SL bf16
  bf16_t* xk_bf  = xr_bf + SL;                       // SL bf16
  bf16_t* xv_bf  = (bf16_t*)(ws + 12 * SL);          // SL bf16
  bf16_t* wgt_bf = xv_bf + SL;                       // WN bf16 (8 MB)
  bf16_t* yg_bf  = xr_bf;                            // reuse (gemm_r long done)
  // float-slot reuse after their consumers finish
  float* kkb = xw;  float* bbb = xa;  float* yb = xg;
  float* outp = (float*)d_out;

  dim3 gemmGrid(M / 128, N / 256), gemmBlk(256);
  const unsigned cvtW = (unsigned)(WN / (8 * 256));  // weight convert grid

  // 1) token shift + mixes (bf16 GEMM operands emitted directly)
  mix_kernel<<<(unsigned)(SL / 256), 256, 0, stream>>>(
      x, c_r, c_w, c_k, c_v, c_a, c_g, xr_bf, xw, xk_bf, xv, xv_bf, xa, xg);

  // 2) LoRA stage-1 (xw/xa/xg consumed here; xv float kept for v-LoRA)
  lora_stage1<<<M, 64, 0, stream>>>(xw, w1, tmp_w, 64, 1);    // tanh
  lora_stage1<<<M, 64, 0, stream>>>(xa, a1, tmp_a, 64, 0);
  lora_stage1<<<M, 32, 0, stream>>>(xv, v1, tmp_v, 32, 0);
  lora_stage1<<<M, 128, 0, stream>>>(xg, g1, tmp_g, 128, 2);  // sigmoid

  // 3) big projections on WMMA (convert weight, then NT gemm: X @ W.T)
  cvt_f32_bf16<<<cvtW, 256, 0, stream>>>(W_r, wgt_bf);
  gemm_nt_wmma<<<gemmGrid, gemmBlk, 0, stream>>>(xr_bf, wgt_bf, rb, M, N, K);
  cvt_f32_bf16<<<cvtW, 256, 0, stream>>>(W_k, wgt_bf);
  gemm_nt_wmma<<<gemmGrid, gemmBlk, 0, stream>>>(xk_bf, wgt_bf, kb, M, N, K);
  cvt_f32_bf16<<<cvtW, 256, 0, stream>>>(W_v, wgt_bf);
  gemm_nt_wmma<<<gemmGrid, gemmBlk, 0, stream>>>(xv_bf, wgt_bf, vb, M, N, K);

  // 4) LoRA stage-2 epilogues
  dim3 l2grid(M, Csz / 256);
  lora_stage2<<<l2grid, 256, 0, stream>>>(tmp_w, w2, w0, 64, 0, nullptr, nullptr, wb);  // wdec
  lora_stage2<<<l2grid, 256, 0, stream>>>(tmp_a, a2, a0, 64, 1, nullptr, nullptr, ab);  // a
  lora_stage2<<<l2grid, 256, 0, stream>>>(tmp_v, v2, v0, 32, 2, vb, vfirst, vb);        // v blend
  lora_stage2<<<l2grid, 256, 0, stream>>>(tmp_g, g2, nullptr, 128, 3, nullptr, nullptr, gb); // g

  // 5) kk normalize + derived sequences (k updated in place)
  int groups = Bsz * Tsz * Hh;
  kknorm_kernel<<<groups / 8, 256, 0, stream>>>(kb, ab, k_k, k_a, kkb, bbb, kb);

  // 6) sequential delta-rule recurrence
  rwkv_rec_kernel<<<Bsz * Hh, 256, 0, stream>>>(rb, wb, kb, vb, kkb, bbb, yb);

  // 7) groupnorm + rk*v residual + gate -> bf16
  post_kernel<<<groups / 8, 256, 0, stream>>>(yb, rb, kb, vb, gb, r_k, lng, lnb, yg_bf);

  // 8) output projection
  cvt_f32_bf16<<<cvtW, 256, 0, stream>>>(W_o, wgt_bf);
  gemm_nt_wmma<<<gemmGrid, gemmBlk, 0, stream>>>(yg_bf, wgt_bf, outp, M, N, K);

  // 9) second tuple element: pass-through v_first
  hipMemcpyAsync(outp + SL, vfirst, SL * sizeof(float),
                 hipMemcpyDeviceToDevice, stream);
}
